// SpatialAttentionBlock_2353642078870
// MI455X (gfx1250) — matrix operand order repaired and checked by reading, not compiled, not run on hardware
//
#include <hip/hip_runtime.h>

// SpatialAttentionBlock for MI455X (gfx1250, wave32, WMMA).
// B=4, C=64, Cq=8, H=W=64, N=4096.
//
// Transposed flash attention:
//   S^T = K_tile x Q        (v_wmma_f32_16x16x32_f16, K=8 zero-padded)
//   out^T = V_tile x P^T    (v_wmma_f32_16x16x32_f16, K=32 exact)
// Softmax uses a static per-row shift M_i = q_i . kmax (valid since q,k >= 0
// post-ReLU => S <= M elementwise), computed in log2 domain with a +14 bias
// so f16 P stays in [2^-24, 2^14]. exp is raw v_exp_f32. The Q operand is
// zero for all K >= 8, so the K-tile A operand needs no lane masking at all.

#define BB   4
#define CC   64
#define CQ   8
#define HH   64
#define WW   64
#define NPIX (HH * WW)
#define LOG2E 1.44269504f

typedef __attribute__((ext_vector_type(16))) _Float16     v16h;
typedef __attribute__((ext_vector_type(2)))  __fp16       v2fp16;
typedef __attribute__((ext_vector_type(8)))  float        v8f;
typedef __attribute__((ext_vector_type(4)))  unsigned int u32x4;

union V16 { v16h h; u32x4 u4[2]; unsigned int u[8]; };

static __device__ __forceinline__ unsigned int pack2h(float a, float b) {
    union { v2fp16 h; unsigned int u; } t;
    t.h = __builtin_amdgcn_cvt_pkrtz(a, b);   // one v_cvt_pk_rtz_f16_f32
    return t.u;
}

// ---------------------------------------------------------------------------
// Kernel 0: zero the per-(batch,channel) k-max accumulators (32 floats).
// ---------------------------------------------------------------------------
__global__ void kmax_init_kernel(float* __restrict__ kmax) {
    kmax[threadIdx.x] = 0.f;   // 0.0f bits == 0; k >= 0 so 0 is the identity
}

// ---------------------------------------------------------------------------
// Kernel 1: fused q/k/v projections.
//   q = relu(bn(conv1x3(x))) * log2(e)   -> qf  pixel-major [b][pix][8] f16
//   k = relu(bn(conv3x1(x)))             -> kf  pixel-major [b][pix][8] f16
//   v = conv1x1(x)                       -> vf  channel-major [b][c][pix] f16
// Also reduces kmax[b][c] = max_pix k (over the f16-rounded values).
// ---------------------------------------------------------------------------
__global__ __launch_bounds__(256) void qkv_proj_kernel(
    const float* __restrict__ x,
    const float* __restrict__ qw,   const float* __restrict__ qb,
    const float* __restrict__ qbnw, const float* __restrict__ qbnb,
    const float* __restrict__ qbnm, const float* __restrict__ qbnv,
    const float* __restrict__ kw,   const float* __restrict__ kbias,
    const float* __restrict__ kbnw, const float* __restrict__ kbnb,
    const float* __restrict__ kbnm, const float* __restrict__ kbnv,
    const float* __restrict__ vw,   const float* __restrict__ vb,
    _Float16* __restrict__ qf, _Float16* __restrict__ kf,
    _Float16* __restrict__ vf, float* __restrict__ kmaxg)
{
    __shared__ int skmax[CQ];
    if (threadIdx.x < CQ) skmax[threadIdx.x] = 0;
    __syncthreads();

    const int pix = blockIdx.x * 256 + threadIdx.x;     // 0 .. B*N-1
    const int b = pix >> 12;                            // block is single-b
    const int i = pix & (NPIX - 1);
    const int h = i >> 6;
    const int w = i & (WW - 1);
    const float* xb = x + (size_t)b * CC * NPIX + i;

    float qacc[CQ], kacc[CQ], vacc[CC];
#pragma unroll
    for (int c = 0; c < CQ; ++c) { qacc[c] = 0.f; kacc[c] = 0.f; }
#pragma unroll
    for (int c = 0; c < CC; ++c) vacc[c] = 0.f;

    for (int ci = 0; ci < CC; ++ci) {
        const float* xc = xb + ci * NPIX;
        const float xm = xc[0];
        const float xl = (w > 0)      ? xc[-1]  : 0.f;   // 1x3 pad (0,1)
        const float xr = (w < WW - 1) ? xc[1]   : 0.f;
        const float xu = (h > 0)      ? xc[-WW] : 0.f;   // 3x1 pad (1,0)
        const float xd = (h < HH - 1) ? xc[WW]  : 0.f;
#pragma unroll
        for (int c = 0; c < CQ; ++c) {
            const float* qwp = qw + (c * CC + ci) * 3;   // uniform -> s_load
            const float* kwp = kw + (c * CC + ci) * 3;
            qacc[c] = fmaf(xl, qwp[0], fmaf(xm, qwp[1], fmaf(xr, qwp[2], qacc[c])));
            kacc[c] = fmaf(xu, kwp[0], fmaf(xm, kwp[1], fmaf(xd, kwp[2], kacc[c])));
        }
#pragma unroll
        for (int c = 0; c < CC; ++c)
            vacc[c] = fmaf(xm, vw[c * CC + ci], vacc[c]);
    }

#pragma unroll
    for (int c = 0; c < CQ; ++c) {
        const float qi = qbnw[c] * rsqrtf(qbnv[c] + 1e-5f);
        const float qo = qb[c] * qi + qbnb[c] - qbnm[c] * qi;
        const float ki = kbnw[c] * rsqrtf(kbnv[c] + 1e-5f);
        const float ko = kbias[c] * ki + kbnb[c] - kbnm[c] * ki;
        const float qv = fmaxf(qacc[c] * qi + qo, 0.f) * LOG2E;  // log2 domain
        const _Float16 kh = (_Float16)fmaxf(kacc[c] * ki + ko, 0.f);
        qf[((size_t)b * NPIX + i) * CQ + c] = (_Float16)qv;
        kf[((size_t)b * NPIX + i) * CQ + c] = kh;
        // nonneg float bits order like ints -> int atomic max
        atomicMax(&skmax[c], __float_as_int((float)kh));
    }
#pragma unroll
    for (int c = 0; c < CC; ++c)
        vf[((size_t)(b * CC + c)) * NPIX + i] = (_Float16)(vacc[c] + vb[c]);

    __syncthreads();
    if (threadIdx.x < CQ)
        atomicMax((int*)(kmaxg + b * CQ + threadIdx.x), skmax[threadIdx.x]);
}

// ---------------------------------------------------------------------------
// Kernel 2: transposed flash attention with static softmax shift.
// 4 waves/block, 16 query columns per wave (64 per block).
// ---------------------------------------------------------------------------
__global__ __launch_bounds__(128) void flash_attn_kernel(
    const _Float16* __restrict__ qf, const _Float16* __restrict__ kf,
    const _Float16* __restrict__ vf, const float* __restrict__ x,
    const float* __restrict__ gamma, const float* __restrict__ kmaxg,
    float* __restrict__ out)
{
    const int lane = threadIdx.x & 31;
    const int half = lane >> 4;          // 0: lanes 0-15, 1: lanes 16-31
    const int l16  = lane & 15;
    const int wave = threadIdx.x >> 5;
    const int b    = blockIdx.x >> 6;                       // 64 blocks/batch
    const int iblk = ((blockIdx.x & 63) << 6) + (wave << 4);
    const int i    = iblk + l16;         // this lane's query (pixel) column

    const v16h hz = {};
    const v8f  fz = {};

    const _Float16* qbp = qf + (size_t)b * NPIX * CQ;
    const _Float16* kbp = kf + (size_t)b * NPIX * CQ;
    const _Float16* vbp = vf + (size_t)b * CC * NPIX;

    // Q operand (B matrix 32x16): lanes 0-15 col i, halves 0-7 = channels.
    // Zero everywhere K >= 8 -- this masking (done once) is what lets the
    // per-tile K loads below run unmasked on all lanes.
    V16 qB; qB.h = hz;
    if (half == 0)
        qB.u4[0] = *(const u32x4*)(qbp + (size_t)i * CQ);

    // Static shift: M = q_i . kmax  (>= every S since q,k >= 0; both sides
    // built from the same f16 values). Bias +14 keeps f16 P <= 2^14.
    V16 qrow; qrow.h = hz;
    qrow.u4[0] = *(const u32x4*)(qbp + (size_t)i * CQ);
    float M = 0.f;
#pragma unroll
    for (int c = 0; c < CQ; ++c)
        M = fmaf((float)qrow.h[c], kmaxg[b * CQ + c], M);   // kmax: s_load
    const float Moff = M - 14.0f;

    float l = 0.f;                       // per-lane partial denominator
    v8f acc[4];
#pragma unroll
    for (int ct = 0; ct < 4; ++ct) acc[ct] = fz;

#pragma unroll 2
    for (int j0 = 0; j0 < NPIX; j0 += 32) {
        if (j0 + 32 < NPIX)   // global_prefetch_b8 for next K tile
            __builtin_prefetch(kbp + (size_t)(j0 + 32 + lane) * CQ, 0, 3);

        // K operands (A matrices 16x32). Unconditional on all lanes: the
        // K>=8 slots are don't-care (multiplied by qB's zeros), so just
        // duplicate the row dword4 instead of masking EXEC.
        V16 ka0, ka1;
        ka0.u4[0] = *(const u32x4*)(kbp + (size_t)(j0 + l16) * CQ);
        ka0.u4[1] = ka0.u4[0];
        ka1.u4[0] = *(const u32x4*)(kbp + (size_t)(j0 + 16 + l16) * CQ);
        ka1.u4[1] = ka1.u4[0];

        v8f st0 = fz, st1 = fz;   // S^T (log2 units): slot r -> j=j0(+16)+r+8*half
        st0 = __builtin_amdgcn_wmma_f32_16x16x32_f16(false, ka0.h, false, qB.h,
                                                     (short)0, st0, false, false);
        st1 = __builtin_amdgcn_wmma_f32_16x16x32_f16(false, ka1.h, false, qB.h,
                                                     (short)0, st1, false, false);

        // P = 2^(S - M + 14): raw v_exp_f32, no libm range fixup.
        float p0[8], p1[8];
#pragma unroll
        for (int r = 0; r < 8; ++r) {
            p0[r] = __builtin_amdgcn_exp2f(st0[r] - Moff);
            p1[r] = __builtin_amdgcn_exp2f(st1[r] - Moff);
            l += p0[r] + p1[r];
        }

        // Build P^T B-operand (32x16): pack f16 pairs in-lane, swap halves.
        // half0 lane owns j-local {0..7}(st0),{16..23}(st1);
        // half1 lane owns j-local {8..15}(st0),{24..31}(st1).
        V16 pB;
#pragma unroll
        for (int v = 0; v < 4; ++v) {
            const unsigned int pk0 = pack2h(p0[2 * v], p0[2 * v + 1]);
            const unsigned int pk1 = pack2h(p1[2 * v], p1[2 * v + 1]);
            const unsigned int x0 = (unsigned int)__shfl_xor((int)pk0, 16, 32);
            const unsigned int x1 = (unsigned int)__shfl_xor((int)pk1, 16, 32);
            pB.u[v]     = half ? x1  : pk0;   // K = 2v,2v+1   (+16 for half1)
            pB.u[4 + v] = half ? pk1 : x0;    // K = 8+2v,...  (+16 for half1)
        }

        // V tiles as A operand (16c x 32j): contiguous b128 loads, all lanes.
#pragma unroll
        for (int ct = 0; ct < 4; ++ct) {
            const _Float16* vp = vbp + (size_t)(ct * 16 + l16) * NPIX
                               + j0 + (half << 4);
            V16 va;
            va.u4[0] = *(const u32x4*)(vp);        // K 0-7  / 16-23
            va.u4[1] = *(const u32x4*)(vp + 8);    // K 8-15 / 24-31
            acc[ct] = __builtin_amdgcn_wmma_f32_16x16x32_f16(
                false, va.h, false, pB.h, (short)0, acc[ct], false, false);
        }
    }

    // Denominator: fold the other half's partial once at the end.
    l += __shfl_xor(l, 16, 32);
    const float g = gamma[0];
    const float inv = 1.f / l;

    // Epilogue: D2 layout row c = ct*16 + r + 8*half, col i = iblk + l16.
    // Loads/stores lane-contiguous along the pixel dimension.
#pragma unroll
    for (int ct = 0; ct < 4; ++ct) {
#pragma unroll
        for (int r = 0; r < 8; ++r) {
            const int c = ct * 16 + r + (half << 3);
            const size_t off = ((size_t)(b * CC + c)) * NPIX + i;
            out[off] = g * (acc[ct][r] * inv) + x[off];
        }
    }
}

// ---------------------------------------------------------------------------
extern "C" void kernel_launch(void* const* d_in, const int* in_sizes, int n_in,
                              void* d_out, int out_size, void* d_ws, size_t ws_size,
                              hipStream_t stream)
{
    const float* x     = (const float*)d_in[0];
    const float* q_w   = (const float*)d_in[1];
    const float* q_b   = (const float*)d_in[2];
    const float* q_bnw = (const float*)d_in[3];
    const float* q_bnb = (const float*)d_in[4];
    const float* q_bnm = (const float*)d_in[5];
    const float* q_bnv = (const float*)d_in[6];
    const float* k_w   = (const float*)d_in[7];
    const float* k_b   = (const float*)d_in[8];
    const float* k_bnw = (const float*)d_in[9];
    const float* k_bnb = (const float*)d_in[10];
    const float* k_bnm = (const float*)d_in[11];
    const float* k_bnv = (const float*)d_in[12];
    const float* v_w   = (const float*)d_in[13];
    const float* v_b   = (const float*)d_in[14];
    const float* gamma = (const float*)d_in[15];
    float* out = (float*)d_out;

    // Workspace: qf/kf [B][N][8] f16, vf [B][64][N] f16, kmax [B][8] f32
    _Float16* qf = (_Float16*)d_ws;
    _Float16* kf = qf + (size_t)BB * NPIX * CQ;
    _Float16* vf = kf + (size_t)BB * NPIX * CQ;
    float* kmaxws = (float*)(vf + (size_t)BB * CC * NPIX);

    kmax_init_kernel<<<1, BB * CQ, 0, stream>>>(kmaxws);

    qkv_proj_kernel<<<(BB * NPIX) / 256, 256, 0, stream>>>(
        x, q_w, q_b, q_bnw, q_bnb, q_bnm, q_bnv,
        k_w, k_b, k_bnw, k_bnb, k_bnm, k_bnv,
        v_w, v_b, qf, kf, vf, kmaxws);

    flash_attn_kernel<<<BB * (NPIX / 64), 128, 0, stream>>>(
        qf, kf, vf, x, gamma, kmaxws, out);
}
